// MultiHeadAttention_43181601194320
// MI455X (gfx1250) — matrix operand order repaired and checked by reading, not compiled, not run on hardware
//
#include <hip/hip_runtime.h>
#include <stdint.h>

// MI455X / gfx1250: wave32, WMMA bf16 16x16x32 with f32 accumulate.

typedef __bf16 v16bf __attribute__((ext_vector_type(16)));
typedef __bf16 v2bf  __attribute__((ext_vector_type(2)));
typedef float  v8f   __attribute__((ext_vector_type(8)));

#define DM    512
#define NH    8
#define DEPTH 64
#define BB    16
#define SS    1024
#define PST   1032   // LDS probs row stride (ushorts), padded vs 1024 to avoid bank conflicts

// Hardware bf16 conversion (v_cvt_*_bf16_f32 path) instead of manual integer RNE.
__device__ __forceinline__ uint16_t f2bf(float x) {
    return __builtin_bit_cast(uint16_t, (__bf16)x);
}
__device__ __forceinline__ uint32_t pack2bf(float lo, float hi) {
    v2bf v = { (__bf16)lo, (__bf16)hi };
    return __builtin_bit_cast(uint32_t, v);
}

union Frag { v16bf bf; uint32_t u[8]; };

// ---- Fragment loaders, contiguous k = 16*hi + s in both A and B, not the ISA table order (wave32) ----
// A (16x32 MxK): lane L holds row M=L&15; lanes<16: K=0..15; lanes>=16: K=16..31 (dw j -> K=16*hi+2j,2j+1).
// 32B contiguous per lane.
__device__ __forceinline__ v16bf frag_a_bf16(const uint16_t* A, int lda, int row0, int k0) {
    int lane = threadIdx.x & 31, hi = (lane >> 4) & 1;
    const uint16_t* base = A + (size_t)(row0 + (lane & 15)) * lda + k0 + hi * 16;
    Frag f;
#pragma unroll
    for (int j = 0; j < 4; ++j) f.u[j]     = *(const uint32_t*)(base + 2*j);
#pragma unroll
    for (int j = 0; j < 4; ++j) f.u[4 + j] = *(const uint32_t*)(base + 8 + 2*j);
    return f.bf;
}
// B (32x16 KxN) sourced from a row-major [N][K] bf16 matrix (implicit transpose: C = A * W^T).
// lane L holds column N=L&15; lanes<16: K=0..15; lanes>=16: K=16..31. 32B contiguous per lane.
__device__ __forceinline__ v16bf frag_b_bf16(const uint16_t* W, int ldb, int n0, int k0) {
    int lane = threadIdx.x & 31, hi = (lane >> 4) & 1;
    const uint16_t* base = W + (size_t)(n0 + (lane & 15)) * ldb + k0 + hi * 16;
    Frag f;
#pragma unroll
    for (int j = 0; j < 8; ++j) f.u[j] = *(const uint32_t*)(base + 2*j);
    return f.bf;
}
// B (32x16 KxN) sourced from a row-major [K][N] bf16 matrix (V: keys x depth) -> strided gather.
__device__ __forceinline__ v16bf frag_b_kmajor_bf16(const uint16_t* V, int ldn, int n0, int k0) {
    int lane = threadIdx.x & 31, hi = (lane >> 4) & 1;
    int n = n0 + (lane & 15);
    int kb = k0 + hi * 16;
    Frag f;
#pragma unroll
    for (int j = 0; j < 8; ++j) {
        uint32_t lo  = V[(size_t)(kb + 2*j)     * ldn + n];
        uint32_t hi2 = V[(size_t)(kb + 2*j + 1) * ldn + n];
        f.u[j] = lo | (hi2 << 16);
    }
    return f.bf;
}

__device__ __forceinline__ v8f wmma_bf16(v16bf a, v16bf b, v8f c) {
    return __builtin_amdgcn_wmma_f32_16x16x32_bf16(false, a, false, b, (short)0, c, false, false);
}

// ---------------- Kernel 0: one-time f32 -> bf16 conversion (vectorized) ----------------
__global__ __launch_bounds__(256) void cvt_kernel(const float* __restrict__ src,
                                                  uint16_t* __restrict__ dst, int n4) {
    int i = blockIdx.x * 256 + threadIdx.x;
    if (i >= n4) return;
    float4 f = ((const float4*)src)[i];
    uint2 o;
    o.x = pack2bf(f.x, f.y);
    o.y = pack2bf(f.z, f.w);
    ((uint2*)dst)[i] = o;
}

// ---------------- Kernel 1: fused QKV projections -> head-split bf16 ----------------
// block = 16 rows x 512 cols (8 waves x 4 N-tiles), grid.y selects q/k/v
__global__ __launch_bounds__(256) void proj_kernel(
    const uint16_t* __restrict__ qb, const uint16_t* __restrict__ kb, const uint16_t* __restrict__ vb,
    const uint16_t* __restrict__ wqb, const uint16_t* __restrict__ wkb, const uint16_t* __restrict__ wvb,
    const float* __restrict__ bq, const float* __restrict__ bk, const float* __restrict__ bv,
    uint16_t* __restrict__ Qh, uint16_t* __restrict__ Kh, uint16_t* __restrict__ Vh)
{
    int sel = blockIdx.y;
    const uint16_t* A    = sel == 0 ? qb  : sel == 1 ? kb  : vb;
    const uint16_t* W    = sel == 0 ? wqb : sel == 1 ? wkb : wvb;
    const float*    bias = sel == 0 ? bq  : sel == 1 ? bk  : bv;
    uint16_t*       Out  = sel == 0 ? Qh  : sel == 1 ? Kh  : Vh;

    int row0 = blockIdx.x * 16;
    int w = threadIdx.x >> 5, lane = threadIdx.x & 31;
    int hi = (lane >> 4) & 1, l15 = lane & 15;
    int n0 = w * 64;

    v8f c[4] = {};
    for (int k0 = 0; k0 < DM; k0 += 32) {
        v16bf a = frag_a_bf16(A, DM, row0, k0);
#pragma unroll
        for (int t = 0; t < 4; ++t)
            c[t] = wmma_bf16(a, frag_b_bf16(W, DM, n0 + 16 * t, k0), c[t]);
    }
#pragma unroll
    for (int t = 0; t < 4; ++t) {
        int col = n0 + 16 * t + l15;
        int h = col >> 6, d = col & 63;
        float bval = bias[col];
#pragma unroll
        for (int r = 0; r < 8; ++r) {
            int row = row0 + r + hi * 8;        // C/D layout: VGPR r -> M = r + 8*hi
            int b = row >> 10, s = row & 1023;
            Out[(((size_t)(b * NH + h)) * SS + s) * DEPTH + d] = f2bf(c[t][r] + bval);
        }
    }
}

// ---------------- Kernel 2: fused attention (logits+softmax+PV) ----------------
// block = one (b, h, 16-query tile); 256 threads = 8 waves
__global__ __launch_bounds__(256) void attn_kernel(
    const uint16_t* __restrict__ Qh, const uint16_t* __restrict__ Kh, const uint16_t* __restrict__ Vh,
    const int* __restrict__ mask, const float* __restrict__ adjoin, const float* __restrict__ dist,
    float* __restrict__ attnW, uint16_t* __restrict__ AO)
{
    extern __shared__ char smem[];
    float*    scores = (float*)smem;                                    // [16][1024] f32
    uint16_t* probs  = (uint16_t*)(smem + 16 * 1024 * 4);               // [16][PST] bf16
    float*    red    = (float*)(smem + 16 * 1024 * 4 + 16 * PST * 2);   // [16][16]
    float*    rowmax = red + 256;                                       // [16]
    float*    rowsum = rowmax + 16;                                     // [16]
    float*    obuf   = rowsum + 16;                                     // [16][64]

    int b = blockIdx.z, h = blockIdx.y;
    int q0 = blockIdx.x * 16;
    int tid = threadIdx.x;
    int w = tid >> 5, lane = tid & 31, hi = (lane >> 4) & 1, l15 = lane & 15;

    const uint16_t* Qp = Qh + (((size_t)(b * NH + h)) * SS + q0) * DEPTH;
    const uint16_t* Kp = Kh + ((size_t)(b * NH + h)) * SS * DEPTH;
    const uint16_t* Vp = Vh + ((size_t)(b * NH + h)) * SS * DEPTH;
    size_t mbase = (size_t)b * SS * SS;

    // Q tile fragments are reused across all 64 key tiles
    v16bf qf0 = frag_a_bf16(Qp, DEPTH, 0, 0);
    v16bf qf1 = frag_a_bf16(Qp, DEPTH, 0, 32);

    // ---- Phase 1: logits = relu(QK^T)*rescale(dist)/8 + mask*(-1e9) + adjoin ----
    for (int t = 0; t < 8; ++t) {
        int n0k = (w * 8 + t) * 16;
        v8f c = {};
        c = wmma_bf16(qf0, frag_b_bf16(Kp, DEPTH, n0k, 0),  c);
        c = wmma_bf16(qf1, frag_b_bf16(Kp, DEPTH, n0k, 32), c);
        int n = n0k + l15;
#pragma unroll
        for (int r = 0; r < 8; ++r) {
            int rloc = r + hi * 8;
            size_t gi = mbase + (size_t)(q0 + rloc) * SS + n;
            float dv = dist[gi];
            float resc = 3.7182818f / (1.f + __expf(1.f - dv));  // (1+e)/(1+exp(1-w))
            float val = fmaxf(c[r], 0.f) * resc * 0.125f;        // /sqrt(64)
            val += (float)mask[gi] * -1e9f + adjoin[gi];
            scores[rloc * 1024 + n] = val;
        }
    }
    __syncthreads();

    // ---- Phase 2: row softmax over 1024 keys ----
    int row = tid >> 4, c16 = tid & 15;
    float m = -3.4e38f;
    for (int j = 0; j < 64; ++j) m = fmaxf(m, scores[row * 1024 + c16 + j * 16]);
    red[row * 16 + c16] = m;
    __syncthreads();
    if (c16 == 0) {
        float mm = red[row * 16];
        for (int j = 1; j < 16; ++j) mm = fmaxf(mm, red[row * 16 + j]);
        rowmax[row] = mm;
    }
    __syncthreads();
    float rm = rowmax[row];
    float sum = 0.f;
    for (int j = 0; j < 64; ++j) {
        int col = c16 + j * 16;
        float e = __expf(scores[row * 1024 + col] - rm);
        scores[row * 1024 + col] = e;
        sum += e;
    }
    red[row * 16 + c16] = sum;
    __syncthreads();
    if (c16 == 0) {
        float s2 = 0.f;
        for (int j = 0; j < 16; ++j) s2 += red[row * 16 + j];
        rowsum[row] = 1.f / s2;      // store reciprocal
    }
    __syncthreads();

    // Coalesced prob write: 256 consecutive floats per iteration (biggest HBM stream).
    float* attnBase = attnW + (((size_t)(b * NH + h)) * SS + q0) * SS;
    for (int e = tid; e < 16 * 1024; e += 256) {
        int r = e >> 10, col = e & 1023;
        float p = scores[e] * rowsum[r];
        attnBase[e] = p;                     // required second output (f32)
        probs[r * PST + col] = f2bf(p);      // bf16 operand for P*V
    }
    for (int e = tid; e < 16 * 64; e += 256) obuf[e] = 0.f;
    __syncthreads();

    // ---- Phase 3: O = P[16x1024] x V[1024x64]; 8 waves = 4 N-tiles x 2 K-halves ----
    int nt = w & 3, khalf = w >> 2;
    v8f oc = {};
    for (int k0 = khalf * 512; k0 < khalf * 512 + 512; k0 += 32) {
        v16bf pa = frag_a_bf16(probs, PST, 0, k0);
        v16bf vfr = frag_b_kmajor_bf16(Vp, DEPTH, nt * 16, k0);
        oc = wmma_bf16(pa, vfr, oc);
    }
#pragma unroll
    for (int r = 0; r < 8; ++r)
        atomicAdd(&obuf[(r + hi * 8) * 64 + nt * 16 + l15], oc[r]);  // ds_add_f32 combine
    __syncthreads();

    // store merged-head bf16 [B,S,512] for the dense projection
    for (int e = tid; e < 16 * 64; e += 256) {
        int rr = e >> 6, d = e & 63;
        AO[((size_t)(b * SS + q0 + rr)) * DM + h * DEPTH + d] = f2bf(obuf[e]);
    }
}

// ---------------- Kernel 3: output dense projection ----------------
__global__ __launch_bounds__(256) void dense_kernel(
    const uint16_t* __restrict__ AO, const uint16_t* __restrict__ Wd,
    const float* __restrict__ bd, float* __restrict__ out)
{
    int row0 = blockIdx.x * 16;
    int w = threadIdx.x >> 5, lane = threadIdx.x & 31;
    int hi = (lane >> 4) & 1, l15 = lane & 15;
    int n0 = w * 64;
    v8f c[4] = {};
    for (int k0 = 0; k0 < DM; k0 += 32) {
        v16bf a = frag_a_bf16(AO, DM, row0, k0);
#pragma unroll
        for (int t = 0; t < 4; ++t)
            c[t] = wmma_bf16(a, frag_b_bf16(Wd, DM, n0 + 16 * t, k0), c[t]);
    }
#pragma unroll
    for (int t = 0; t < 4; ++t) {
        int col = n0 + 16 * t + l15;
        float bval = bd[col];
#pragma unroll
        for (int r = 0; r < 8; ++r) {
            int rowg = row0 + r + hi * 8;
            out[(size_t)rowg * DM + col] = c[t][r] + bval;
        }
    }
}

extern "C" void kernel_launch(void* const* d_in, const int* in_sizes, int n_in,
                              void* d_out, int out_size, void* d_ws, size_t ws_size,
                              hipStream_t stream) {
    const float* q      = (const float*)d_in[0];
    const float* k      = (const float*)d_in[1];
    const float* v      = (const float*)d_in[2];
    const int*   mask   = (const int*)  d_in[3];
    const float* adjoin = (const float*)d_in[4];
    const float* dist   = (const float*)d_in[5];
    const float* wq     = (const float*)d_in[6];
    const float* bq     = (const float*)d_in[7];
    const float* wk     = (const float*)d_in[8];
    const float* bk     = (const float*)d_in[9];
    const float* wv     = (const float*)d_in[10];
    const float* bv     = (const float*)d_in[11];
    const float* wd     = (const float*)d_in[12];
    const float* bd     = (const float*)d_in[13];

    float* out   = (float*)d_out;                       // [B,S,512]
    float* attnW = out + (size_t)BB * SS * DM;          // [B,H,S,S]

    const size_t NTOK = (size_t)BB * SS * DM;           // 8,388,608 elements
    const size_t NW   = (size_t)DM * DM;                // 262,144 elements

    uint16_t* Qh  = (uint16_t*)d_ws;                    // bf16 [B,H,S,64]
    uint16_t* Kh  = Qh  + NTOK;
    uint16_t* Vh  = Kh  + NTOK;
    uint16_t* AO  = Vh  + NTOK;                         // bf16 [B,S,512]
    uint16_t* qb  = AO  + NTOK;                         // bf16 copies of inputs
    uint16_t* kb  = qb  + NTOK;
    uint16_t* vb  = kb  + NTOK;
    uint16_t* wqb = vb  + NTOK;                         // bf16 weights
    uint16_t* wkb = wqb + NW;
    uint16_t* wvb = wkb + NW;
    uint16_t* wdb = wvb + NW;

    // one-time conversions
    int n4t = (int)(NTOK / 4), n4w = (int)(NW / 4);
    cvt_kernel<<<(n4t + 255) / 256, 256, 0, stream>>>(q,  qb,  n4t);
    cvt_kernel<<<(n4t + 255) / 256, 256, 0, stream>>>(k,  kb,  n4t);
    cvt_kernel<<<(n4t + 255) / 256, 256, 0, stream>>>(v,  vb,  n4t);
    cvt_kernel<<<(n4w + 255) / 256, 256, 0, stream>>>(wq, wqb, n4w);
    cvt_kernel<<<(n4w + 255) / 256, 256, 0, stream>>>(wk, wkb, n4w);
    cvt_kernel<<<(n4w + 255) / 256, 256, 0, stream>>>(wv, wvb, n4w);
    cvt_kernel<<<(n4w + 255) / 256, 256, 0, stream>>>(wd, wdb, n4w);

    dim3 gproj(BB * SS / 16, 3);
    proj_kernel<<<gproj, 256, 0, stream>>>(qb, kb, vb, wqb, wkb, wvb, bq, bk, bv, Qh, Kh, Vh);

    size_t smem = 16 * 1024 * 4        // scores
                + 16 * PST * 2         // probs bf16
                + (256 + 16 + 16 + 16 * 64) * 4;  // red + rowmax + rowsum + obuf
    dim3 gattn(SS / 16, NH, BB);
    attn_kernel<<<gattn, 256, smem, stream>>>(Qh, Kh, Vh, mask, adjoin, dist, attnW, AO);

    dense_kernel<<<BB * SS / 16, 256, 0, stream>>>(AO, wd ? wdb : wdb, bd, out);
}
